// GATRelation_91061896610340
// MI455X (gfx1250) — matrix operand order repaired and checked by reading, not compiled, not run on hardware
//
#include <hip/hip_runtime.h>
#include <math.h>

// ---------------------------------------------------------------------------
// GAT layer, fused for MI455X (gfx1250, wave32, WMMA bf16).
// L=4096, D_IN=512, D_OUT=256, H=4, HEAD=64.
// ---------------------------------------------------------------------------

#define L_DIM   4096
#define D_IN    512
#define D_OUT   256
#define N_HEAD  4
#define HEAD_D  64

typedef __attribute__((ext_vector_type(16))) __bf16        v16bf;
typedef __attribute__((ext_vector_type(8)))  float         v8f;
typedef __attribute__((ext_vector_type(16))) unsigned short v16us;
typedef __attribute__((ext_vector_type(8)))  unsigned short v8us;
typedef __attribute__((ext_vector_type(8)))  int            v8i;

union V16cvt { v16us us; v16bf bf; };

__device__ __forceinline__ unsigned short f2bf(float f) {
    union { float f; unsigned u; } x; x.f = f;
    unsigned r = x.u + 0x7FFFu + ((x.u >> 16) & 1u);   // round-to-nearest-even
    return (unsigned short)(r >> 16);
}

// A-matrix tile loader (16x32 bf16, row-major source, row = rowbase+(lane&15)).
// Per ISA layout: lanes 0-15 hold K {0..7, 16..23}; lanes 16-31 hold K {8..15, 24..31}.
__device__ __forceinline__ v16bf load_a_tile(const unsigned short* __restrict__ prow,
                                             int lane) {
    const int klo = (lane & 16) ? 8 : 0;
    v8us lo = *(const v8us*)(prow + klo);
    v8us hi = *(const v8us*)(prow + 16 + klo);
    V16cvt t;
#pragma unroll
    for (int e = 0; e < 8; ++e) { t.us[e] = lo[e]; t.us[e + 8] = hi[e]; }
    return t.bf;
}

// B-matrix tile loader (32x16 bf16). Source stored N-major (row n, contiguous K).
// Lanes 0-15: N=lane, K {0..7, 16..23}; lanes 16-31: N=lane-16, K {8..15, 24..31}.
__device__ __forceinline__ v16bf load_b_tile(const unsigned short* __restrict__ pnrow,
                                             int lane) {
    const int klo = (lane & 16) ? 8 : 0;
    v8us lo = *(const v8us*)(pnrow + klo);
    v8us hi = *(const v8us*)(pnrow + 16 + klo);
    V16cvt t;
#pragma unroll
    for (int e = 0; e < 8; ++e) { t.us[e] = lo[e]; t.us[e + 8] = hi[e]; }
    return t.bf;
}

// ---------------------------------------------------------------------------
// Kernel 0: f32 -> bf16 conversion
// ---------------------------------------------------------------------------
__global__ void cvt_f32_bf16(const float* __restrict__ src,
                             unsigned short* __restrict__ dst, int n) {
    int i = blockIdx.x * blockDim.x + threadIdx.x;
    if (i < n) dst[i] = f2bf(src[i]);
}

// ---------------------------------------------------------------------------
// Kernel 1: Wh = h @ W^T   (4096x512 @ 512x256), one wave per 16x16 tile.
// Writes Wh (f32, row-major) and WhT (bf16, transposed 256x4096, so that the
// alpha@Wh B-tiles later load contiguously along K=j).
// ---------------------------------------------------------------------------
__global__ __launch_bounds__(32) void gemm_wh_kernel(
    const unsigned short* __restrict__ hb,   // 4096x512 bf16
    const unsigned short* __restrict__ Wb,   // 256x512 bf16 (row n, contig k)
    float* __restrict__ Wh,                  // 4096x256 f32
    unsigned short* __restrict__ WhT) {      // 256x4096 bf16
    const int lane = threadIdx.x;
    const int rowbase = blockIdx.x * 16;
    const int nbase   = blockIdx.y * 16;
    const int row  = rowbase + (lane & 15);
    const int ncol = nbase   + (lane & 15);
    v8f acc = {};
    for (int kb = 0; kb < D_IN; kb += 32) {
        v16bf a = load_a_tile(hb + (size_t)row  * D_IN + kb, lane);
        v16bf b = load_b_tile(Wb + (size_t)ncol * D_IN + kb, lane);
        acc = __builtin_amdgcn_wmma_f32_16x16x32_bf16(false, a, false, b,
                                                      (short)0, acc, false, false);
    }
    const int rbase2 = rowbase + ((lane >> 4) << 3);
    const int nl = lane & 15;
#pragma unroll
    for (int r = 0; r < 8; ++r) {
        const int gr = rbase2 + r;
        const float v = acc[r];
        Wh [(size_t)gr * D_OUT + nbase + nl] = v;
        WhT[(size_t)(nbase + nl) * L_DIM + gr] = f2bf(v);
    }
}

// ---------------------------------------------------------------------------
// Kernel 2: s_i / s_j per-head 64-dot products. One thread per (l, head).
// s1: [4096][4], s2q: [4096][4] (for stats), s2T: [4][4096] (for alpha tiles).
// ---------------------------------------------------------------------------
__global__ void s_kernel(const float* __restrict__ Wh,
                         const float* __restrict__ attn,   // 4x128
                         float* __restrict__ s1, float* __restrict__ s2q,
                         float* __restrict__ s2T) {
    int t = blockIdx.x * blockDim.x + threadIdx.x;
    if (t >= L_DIM * N_HEAD) return;
    const int l = t >> 2, hd = t & 3;
    const float* w  = Wh + (size_t)l * D_OUT + hd * HEAD_D;
    const float* a1 = attn + hd * 2 * HEAD_D;
    const float* a2 = a1 + HEAD_D;
    float x1 = 0.f, x2 = 0.f;
#pragma unroll 8
    for (int d = 0; d < HEAD_D; ++d) { float v = w[d]; x1 += v * a1[d]; x2 += v * a2[d]; }
    s1[t] = x1; s2q[t] = x2; s2T[(size_t)hd * L_DIM + l] = x2;
}

// ---------------------------------------------------------------------------
// Kernel 3: masked online-softmax statistics per (row, head): m and 1/denom.
// One block per row i; threads stream A[i,:] (the 64MB dominant stream),
// maintain per-thread (m,s) for 4 heads, then LDS tree-combine.
// ---------------------------------------------------------------------------
__global__ __launch_bounds__(256) void stats_kernel(
    const int* __restrict__ A, const float* __restrict__ s1,
    const float* __restrict__ s2q,
    float* __restrict__ mout, float* __restrict__ invd) {
    __shared__ float rm[256][N_HEAD];
    __shared__ float rs[256][N_HEAD];
    const int i = blockIdx.x;
    const int tid = threadIdx.x;
    float s1h[N_HEAD];
#pragma unroll
    for (int h = 0; h < N_HEAD; ++h) s1h[h] = s1[i * N_HEAD + h];
    const int* Arow = A + (size_t)i * L_DIM;
    float mh[N_HEAD] = {-1e30f, -1e30f, -1e30f, -1e30f};
    float sh[N_HEAD] = {0.f, 0.f, 0.f, 0.f};
    for (int j = tid; j < L_DIM; j += 256) {
        __builtin_prefetch(Arow + j + 1024, 0, 0);   // global_prefetch_b8
        if (Arow[j] > 0) {
            const float4 sv = *(const float4*)(s2q + (size_t)j * N_HEAD);
            const float sj[N_HEAD] = {sv.x, sv.y, sv.z, sv.w};
#pragma unroll
            for (int h = 0; h < N_HEAD; ++h) {
                const float x = s1h[h] + sj[h];
                const float e = x > 0.f ? x : 0.2f * x;    // leaky_relu(0.2)
                if (e > mh[h]) { sh[h] = sh[h] * __expf(mh[h] - e) + 1.f; mh[h] = e; }
                else           { sh[h] += __expf(e - mh[h]); }
            }
        }
    }
#pragma unroll
    for (int h = 0; h < N_HEAD; ++h) { rm[tid][h] = mh[h]; rs[tid][h] = sh[h]; }
    for (int off = 128; off > 0; off >>= 1) {
        __syncthreads();
        if (tid < off) {
#pragma unroll
            for (int h = 0; h < N_HEAD; ++h) {
                const float m1 = rm[tid][h],       sa = rs[tid][h];
                const float m2 = rm[tid + off][h], sb = rs[tid + off][h];
                const float m = m1 > m2 ? m1 : m2;
                rm[tid][h] = m;
                rs[tid][h] = sa * __expf(m1 - m) + sb * __expf(m2 - m);
            }
        }
    }
    __syncthreads();
    if (tid < N_HEAD) {
        const float m = rm[0][tid], s = rs[0][tid];
        if (m <= -5e29f) { mout[i * N_HEAD + tid] = 0.f; invd[i * N_HEAD + tid] = 0.f; }
        else             { mout[i * N_HEAD + tid] = m;   invd[i * N_HEAD + tid] = 1.f / s; }
    }
}

// ---------------------------------------------------------------------------
// Kernel 4: fused alpha @ Wh per head. One wave per (16-row block, head).
// Streams j in 32-wide tiles: builds the 16x32 alpha tile directly in the
// WMMA A-register layout (exp folded, 1/denom applied at the end), and
// accumulates 4 N-tiles (HEAD_D=64) with v_wmma_f32_16x16x32_bf16 against
// L2-resident WhT. Avoids ever materializing the 256MB alpha tensor.
// ---------------------------------------------------------------------------
__global__ __launch_bounds__(32) void attn_gemm_kernel(
    const int* __restrict__ A, const float* __restrict__ s1,
    const float* __restrict__ s2T, const float* __restrict__ mbuf,
    const float* __restrict__ invd, const unsigned short* __restrict__ WhT,
    unsigned short* __restrict__ attnb) {             // 4096x256 bf16
    const int lane = threadIdx.x;
    const int rowbase = blockIdx.x * 16;
    const int head = blockIdx.y;
    const int iA = rowbase + (lane & 15);             // A-layout row for this lane
    const float s1i = s1[iA * N_HEAD + head];
    const float mi  = mbuf[iA * N_HEAD + head];
    const int* __restrict__ Arow = A + (size_t)iA * L_DIM;
    const float* __restrict__ s2h = s2T + (size_t)head * L_DIM;
    const int klo = (lane & 16) ? 8 : 0;
    const int ko  = (lane & 16) ? 16 : 0;
    const int nl  = lane & 15;
    const unsigned short* __restrict__ Bbase = WhT + (size_t)(head * HEAD_D) * L_DIM;

    v8f acc0 = {}, acc1 = {}, acc2 = {}, acc3 = {};
    for (int jb = 0; jb < L_DIM; jb += 32) {
        __builtin_prefetch(Arow + jb + 128, 0, 0);
        const v8i m0 = *(const v8i*)(Arow + jb + ko);
        const v8i m1 = *(const v8i*)(Arow + jb + ko + 8);
        const v8f z0 = *(const v8f*)(s2h + jb + ko);
        const v8f z1 = *(const v8f*)(s2h + jb + ko + 8);
        V16cvt av;
#pragma unroll
        for (int t = 0; t < 8; ++t) {
            float x = s1i + z0[t];
            float e = x > 0.f ? x : 0.2f * x;
            av.us[t]     = f2bf((m0[t] > 0) ? __expf(e - mi) : 0.f);
            x = s1i + z1[t];
            e = x > 0.f ? x : 0.2f * x;
            av.us[t + 8] = f2bf((m1[t] > 0) ? __expf(e - mi) : 0.f);
        }
        const v16bf a = av.bf;
        V16cvt b0, b1, b2, b3;
        b0.us = *(const v16us*)(Bbase + (size_t)(nl)      * L_DIM + jb + ko);
        b1.us = *(const v16us*)(Bbase + (size_t)(16 + nl) * L_DIM + jb + ko);
        b2.us = *(const v16us*)(Bbase + (size_t)(32 + nl) * L_DIM + jb + ko);
        b3.us = *(const v16us*)(Bbase + (size_t)(48 + nl) * L_DIM + jb + ko);
        acc0 = __builtin_amdgcn_wmma_f32_16x16x32_bf16(false, a, false, b0.bf, (short)0, acc0, false, false);
        acc1 = __builtin_amdgcn_wmma_f32_16x16x32_bf16(false, a, false, b1.bf, (short)0, acc1, false, false);
        acc2 = __builtin_amdgcn_wmma_f32_16x16x32_bf16(false, a, false, b2.bf, (short)0, acc2, false, false);
        acc3 = __builtin_amdgcn_wmma_f32_16x16x32_bf16(false, a, false, b3.bf, (short)0, acc3, false, false);
    }

    const int rbase2 = rowbase + ((lane >> 4) << 3);   // C/D-layout row base
    float sc[8];
#pragma unroll
    for (int r = 0; r < 8; ++r) sc[r] = invd[(rbase2 + r) * N_HEAD + head];
#pragma unroll
    for (int r = 0; r < 8; ++r) {
        unsigned short* dst = attnb + (size_t)(rbase2 + r) * D_OUT + head * HEAD_D + nl;
        dst[0]  = f2bf(acc0[r] * sc[r]);
        dst[16] = f2bf(acc1[r] * sc[r]);
        dst[32] = f2bf(acc2[r] * sc[r]);
        dst[48] = f2bf(acc3[r] * sc[r]);
    }
}

// ---------------------------------------------------------------------------
// Kernel 5: final = attnout @ out_w^T + out_b  (4096x256 @ 256x256)
// ---------------------------------------------------------------------------
__global__ __launch_bounds__(32) void gemm_out_kernel(
    const unsigned short* __restrict__ attnb,   // 4096x256 bf16
    const unsigned short* __restrict__ out_wb,  // 256x256 bf16 (row n, contig k)
    const float* __restrict__ out_b,
    float* __restrict__ out) {
    const int lane = threadIdx.x;
    const int rowbase = blockIdx.x * 16;
    const int nbase   = blockIdx.y * 16;
    const int row  = rowbase + (lane & 15);
    const int ncol = nbase   + (lane & 15);
    v8f acc = {};
    for (int kb = 0; kb < D_OUT; kb += 32) {
        v16bf a = load_a_tile(attnb  + (size_t)row  * D_OUT + kb, lane);
        v16bf b = load_b_tile(out_wb + (size_t)ncol * D_OUT + kb, lane);
        acc = __builtin_amdgcn_wmma_f32_16x16x32_bf16(false, a, false, b,
                                                      (short)0, acc, false, false);
    }
    const int rbase2 = rowbase + ((lane >> 4) << 3);
    const int nl = lane & 15;
    const float bias = out_b[nbase + nl];
#pragma unroll
    for (int r = 0; r < 8; ++r)
        out[(size_t)(rbase2 + r) * D_OUT + nbase + nl] = acc[r] + bias;
}

// ---------------------------------------------------------------------------
// Host-side launch. Workspace (~12.7 MB) layout, all offsets 256B aligned.
// ---------------------------------------------------------------------------
extern "C" void kernel_launch(void* const* d_in, const int* in_sizes, int n_in,
                              void* d_out, int out_size, void* d_ws, size_t ws_size,
                              hipStream_t stream) {
    const float* h_in  = (const float*)d_in[0];   // 4096x512
    const int*   A     = (const int*)  d_in[1];   // 4096x4096
    const float* W     = (const float*)d_in[2];   // 256x512
    const float* attn  = (const float*)d_in[3];   // 4x128
    const float* out_w = (const float*)d_in[4];   // 256x256
    const float* out_b = (const float*)d_in[5];   // 256
    float* out = (float*)d_out;                   // 4096x256 f32

    char* ws = (char*)d_ws;
    size_t off = 0;
    auto alloc = [&](size_t bytes) { char* p = ws + off; off = (off + bytes + 255) & ~(size_t)255; return p; };
    unsigned short* hb     = (unsigned short*)alloc((size_t)L_DIM * D_IN  * 2);
    unsigned short* Wb     = (unsigned short*)alloc((size_t)D_OUT * D_IN  * 2);
    unsigned short* out_wb = (unsigned short*)alloc((size_t)D_OUT * D_OUT * 2);
    float*          Wh     = (float*)         alloc((size_t)L_DIM * D_OUT * 4);
    unsigned short* WhT    = (unsigned short*)alloc((size_t)D_OUT * L_DIM * 2);
    float*          s1     = (float*)         alloc((size_t)L_DIM * N_HEAD * 4);
    float*          s2q    = (float*)         alloc((size_t)L_DIM * N_HEAD * 4);
    float*          s2T    = (float*)         alloc((size_t)N_HEAD * L_DIM * 4);
    float*          mbuf   = (float*)         alloc((size_t)L_DIM * N_HEAD * 4);
    float*          invd   = (float*)         alloc((size_t)L_DIM * N_HEAD * 4);
    unsigned short* attnb  = (unsigned short*)alloc((size_t)L_DIM * D_OUT * 2);
    (void)ws_size; (void)in_sizes; (void)n_in; (void)out_size;

    // 0) precision conversions
    {
        int n = L_DIM * D_IN;
        cvt_f32_bf16<<<(n + 255) / 256, 256, 0, stream>>>(h_in, hb, n);
        n = D_OUT * D_IN;
        cvt_f32_bf16<<<(n + 255) / 256, 256, 0, stream>>>(W, Wb, n);
        n = D_OUT * D_OUT;
        cvt_f32_bf16<<<(n + 255) / 256, 256, 0, stream>>>(out_w, out_wb, n);
    }
    // 1) Wh = h @ W^T (WMMA bf16), also transposed bf16 copy
    gemm_wh_kernel<<<dim3(L_DIM / 16, D_OUT / 16), 32, 0, stream>>>(hb, Wb, Wh, WhT);
    // 2) per-(row, head) score projections
    s_kernel<<<(L_DIM * N_HEAD + 255) / 256, 256, 0, stream>>>(Wh, attn, s1, s2q, s2T);
    // 3) masked softmax stats (m, 1/denom) over the 64MB adjacency stream
    stats_kernel<<<L_DIM, 256, 0, stream>>>(A, s1, s2q, mbuf, invd);
    // 4) fused alpha @ Wh (WMMA bf16), softmax normalization folded in
    attn_gemm_kernel<<<dim3(L_DIM / 16, N_HEAD), 32, 0, stream>>>(
        A, s1, s2T, mbuf, invd, WhT, attnb);
    // 5) output projection + bias (WMMA bf16)
    gemm_out_kernel<<<dim3(L_DIM / 16, D_OUT / 16), 32, 0, stream>>>(
        attnb, out_wb, out_b, out);
}
